// HybridSTGNN_54924041781321
// MI455X (gfx1250) — matrix operand order repaired and checked by reading, not compiled, not run on hardware
//
#include <hip/hip_runtime.h>
#include <hip/hip_bf16.h>
#include <math.h>

typedef __attribute__((ext_vector_type(16))) _Float16 v16h;
typedef __attribute__((ext_vector_type(8)))  _Float16 v8h;
typedef __attribute__((ext_vector_type(8)))  float    v8f;

// ---------------- constants (match reference) ----------------
#define BB 2
#define WW 8
#define NN 10000
#define FIN 5
#define HH 128
#define BW (BB*WW)          // 16
#define M2 (BB*NN)          // 20000 sequences
#define G4 (4*HH)           // 512 gates

// ---------------- CDNA5 async-to-LDS helpers ----------------
// 32-bit LDS byte offset of a pointer into a __shared__ array.
__device__ __forceinline__ unsigned lds_off(const void* p) {
    return (unsigned)(unsigned long long)(const __attribute__((address_space(3))) char*)p;
}
// GLOBAL_LOAD_ASYNC_TO_LDS_B128: copy 16 contiguous bytes global -> LDS (ASYNCcnt).
__device__ __forceinline__ void async_ld_b128(unsigned lds_byte, const void* gsrc) {
    unsigned long long ga = (unsigned long long)gsrc;
    asm volatile("global_load_async_to_lds_b128 %0, %1, off"
                 :: "v"(lds_byte), "v"(ga)
                 : "memory");
}
__device__ __forceinline__ void wait_async0(void) {
#if __has_builtin(__builtin_amdgcn_s_wait_asynccnt)
    __builtin_amdgcn_s_wait_asynccnt(0);
#else
    asm volatile("s_wait_asynccnt 0x0" ::: "memory");
#endif
}

// ---------------- fragment helper ----------------
// Concatenate two 16-byte (8-half) chunks into one 16-half WMMA fragment.
__device__ __forceinline__ v16h frag_cat(const _Float16* p0, const _Float16* p1) {
    v8h x0 = *(const v8h*)p0;   // ds_load_b128 from LDS-staged tiles
    v8h x1 = *(const v8h*)p1;
    return __builtin_shufflevector(x0, x1, 0, 1, 2, 3, 4, 5, 6, 7,
                                           8, 9, 10, 11, 12, 13, 14, 15);
}

// ---------------- small utility kernels ----------------
__global__ void fill_f32(float* p, int n, float v) {
    int i = blockIdx.x * blockDim.x + threadIdx.x;
    if (i < n) p[i] = v;
}
__global__ void fill_f16(_Float16* p, int n) {
    int i = blockIdx.x * blockDim.x + threadIdx.x;
    if (i < n) p[i] = (_Float16)0.0f;
}
__global__ void conv_f16(const float* __restrict__ in, _Float16* __restrict__ out, int n) {
    int i = blockIdx.x * blockDim.x + threadIdx.x;
    if (i < n) out[i] = (_Float16)in[i];
}
__global__ void degree_k(float* deg, const long long* ei, int E) {
    int e = blockIdx.x * blockDim.x + threadIdx.x;
    if (e < E) atomicAdd(&deg[(int)ei[E + e]], 1.0f);
}
__global__ void rsqrt_k(float* p, int n) {
    int i = blockIdx.x * blockDim.x + threadIdx.x;
    if (i < n) p[i] = rsqrtf(p[i]);
}

// ---------------- GCN layer 1 transform: (BW*N,5) @ (5,128) ----------------
__global__ void gcn1_transform(const float* __restrict__ x,
                               const float* __restrict__ w1,
                               float* __restrict__ ht) {
    int row = blockIdx.x;        // 0..BW*N-1
    int h   = threadIdx.x;       // 0..127
    const float* xr = x + (size_t)row * FIN;
    float acc = 0.0f;
#pragma unroll
    for (int k = 0; k < FIN; ++k) acc = fmaf(xr[k], w1[k * HH + h], acc);
    ht[(size_t)row * HH + h] = acc;
}

// ---------------- GCN aggregate: scatter msg = ht[src]*norm into agg[dst] ----------------
__global__ void gcn_scatter(const float* __restrict__ ht,
                            float* __restrict__ agg,
                            const long long* __restrict__ ei,
                            const float* __restrict__ dinv, int E) {
    int e = blockIdx.x;          // 0..E+N-1  (e >= E: self loop)
    int h = threadIdx.x;         // 0..127
    int s, d;
    if (e < E) { s = (int)ei[e]; d = (int)ei[E + e]; }
    else       { s = e - E; d = s; }
    float nrm = dinv[s] * dinv[d];
#pragma unroll
    for (int bw = 0; bw < BW; ++bw) {
        float v = ht[((size_t)bw * NN + s) * HH + h] * nrm;
        atomicAdd(&agg[((size_t)bw * NN + d) * HH + h], v);
    }
}

// bias + relu epilogue writing f16 (feeds the WMMA GEMMs directly)
__global__ void bias_relu_f16(const float* __restrict__ in,
                              const float* __restrict__ bias,
                              _Float16* __restrict__ out, int total) {
    int i = blockIdx.x * blockDim.x + threadIdx.x;
    if (i < total) {
        float v = in[i] + bias[i & (HH - 1)];
        out[i] = (_Float16)(v > 0.0f ? v : 0.0f);
    }
}

// ---------------- GCN2 WMMA GEMM: C(Mx128) = A(Mx128,f16) @ B(128x128,f16) ----------------
// grid.x = M/16, block = 256 (8 waves, wave w owns 16x16 tile at columns w*16..).
// A strip (16x128) async-copied to LDS (ASYNCcnt) once per block; B staged
// transposed. Padded strides (K+8 halves) keep ds_load_b128 conflict-free.
__global__ __launch_bounds__(256) void gcn2_gemm_wmma(const _Float16* __restrict__ A,
                                                      const _Float16* __restrict__ B,
                                                      float* __restrict__ C) {
    constexpr int K = HH, N = HH, LDK = K + 8;   // 136-half strides
    __shared__ __align__(16) _Float16 bt[N * LDK];   // B transposed: bt[col][k]
    __shared__ __align__(16) _Float16 at[16 * LDK];  // A strip: at[row][k]

    // async-stage A: 16 rows x 16 chunks of 8 halves = 256 chunks, 1 per thread
    {
        int c = threadIdx.x;                 // 0..255
        int row = c >> 4, kc = (c & 15) * 8;
        async_ld_b128(lds_off(&at[row * LDK + kc]),
                      A + (size_t)(blockIdx.x * 16 + row) * K + kc);
    }
    // stage B transposed (coalesced global reads)
    for (int idx = threadIdx.x; idx < K * N; idx += 256) {
        int k = idx >> 7, c = idx & (N - 1);
        bt[c * LDK + k] = B[k * N + c];
    }
    wait_async0();
    __syncthreads();

    const int lane = threadIdx.x & 31;
    const int wave = threadIdx.x >> 5;
    const int lo = lane & 15, hi = lane >> 4;
    const _Float16* ar = at + lo * LDK;
    const _Float16* br = bt + (wave * 16 + lo) * LDK;
    v8f acc = {};
#pragma unroll
    for (int k0 = 0; k0 < K; k0 += 32) {
        v16h a = frag_cat(ar + k0 + hi * 8, ar + k0 + 16 + hi * 8);
        v16h b = frag_cat(br + k0 + hi * 8, br + k0 + 16 + hi * 8);
        acc = __builtin_amdgcn_wmma_f32_16x16x32_f16(false, a, false, b,
                                                     (short)0, acc, false, false);
    }
    const int bcol = wave * 16 + lo;
#pragma unroll
    for (int r = 0; r < 8; ++r) {
        int orow = blockIdx.x * 16 + hi * 8 + r;
        C[(size_t)orow * N + bcol] = acc[r];
    }
}

// ---------------- repack GCN output (BW,N,H) f16 -> Xseq (B*N, W, H) f16 ----------------
__global__ void build_xseq(const _Float16* __restrict__ g2, _Float16* __restrict__ xs) {
    int idx = blockIdx.x * blockDim.x + threadIdx.x;
    if (idx >= M2 * WW * HH) return;
    int h = idx & (HH - 1);
    int rest = idx >> 7;
    int w = rest & (WW - 1);
    int bn = rest >> 3;
    int n = bn % NN, b = bn / NN;
    xs[idx] = g2[(((size_t)(b * WW + w)) * NN + n) * HH + h];
}

// ---------------- pack Wcat(2H x 4H) f16 = [W_ih^T ; W_hh^T], bcat = b_ih + b_hh ----------------
__global__ void build_wcat(const float* __restrict__ wih, const float* __restrict__ whh,
                           const float* __restrict__ bih, const float* __restrict__ bhh,
                           _Float16* __restrict__ wcat, float* __restrict__ bcat) {
    int idx = blockIdx.x * blockDim.x + threadIdx.x;
    if (idx < 2 * HH * G4) {
        int k = idx / G4, j = idx % G4;
        float v = (k < HH) ? wih[(size_t)j * HH + k] : whh[(size_t)j * HH + (k - HH)];
        wcat[idx] = (_Float16)v;
    }
    if (idx < G4) bcat[idx] = bih[idx] + bhh[idx];
}

// ---------------- LSTM gates: (M2 x 512) = [Xt | Hprev](K=256) @ Wcat + bcat ----------------
// grid.x = M2/16, grid.y = 4, block = 256. The 16x256 [Xt|Hprev] A strip is
// async-copied to LDS (concat resolved at staging); Wcat column slice staged
// transposed. Inner K-loop is pure ds_load_b128 + v_wmma.
__global__ __launch_bounds__(256) void lstm_gates_wmma(const _Float16* __restrict__ xs,
                                                       const _Float16* __restrict__ hprev,
                                                       const _Float16* __restrict__ wcat,
                                                       const float* __restrict__ bcat,
                                                       float* __restrict__ gates, int t) {
    constexpr int K = 2 * HH, LDK = K + 8;           // 264-half strides
    __shared__ __align__(16) _Float16 bt[HH * LDK];  // 128 cols x 256 K (transposed)
    __shared__ __align__(16) _Float16 at[16 * LDK];  // 16 rows x 256 K (concat X|H)
    const int colbase = blockIdx.y * HH;

    // async-stage A: 16 rows x 32 chunks of 8 halves = 512 chunks, 2 per thread
#pragma unroll
    for (int c = threadIdx.x; c < 512; c += 256) {
        int row = c >> 5, kc = (c & 31) * 8;
        int arowg = blockIdx.x * 16 + row;
        const _Float16* src = (kc < HH)
            ? xs + ((size_t)arowg * WW + t) * HH + kc
            : hprev + (size_t)arowg * HH + (kc - HH);
        async_ld_b128(lds_off(&at[row * LDK + kc]), src);
    }
    // stage B transposed (coalesced reads of Wcat slice)
    for (int idx = threadIdx.x; idx < K * HH; idx += 256) {
        int k = idx >> 7, c = idx & (HH - 1);
        bt[c * LDK + k] = wcat[(size_t)k * G4 + colbase + c];
    }
    wait_async0();
    __syncthreads();

    const int lane = threadIdx.x & 31;
    const int wave = threadIdx.x >> 5;
    const int lo = lane & 15, hi = lane >> 4;
    const _Float16* ar = at + lo * LDK;
    const _Float16* br = bt + (wave * 16 + lo) * LDK;
    v8f acc = {};
#pragma unroll
    for (int k0 = 0; k0 < K; k0 += 32) {
        v16h a = frag_cat(ar + k0 + hi * 8, ar + k0 + 16 + hi * 8);
        v16h b = frag_cat(br + k0 + hi * 8, br + k0 + 16 + hi * 8);
        acc = __builtin_amdgcn_wmma_f32_16x16x32_f16(false, a, false, b,
                                                     (short)0, acc, false, false);
    }
    const int bcol = colbase + wave * 16 + lo;
    float bb = bcat[bcol];
#pragma unroll
    for (int r = 0; r < 8; ++r) {
        int orow = blockIdx.x * 16 + hi * 8 + r;
        gates[(size_t)orow * G4 + bcol] = acc[r] + bb;
    }
}

__device__ __forceinline__ float sigmoidf(float x) { return 1.0f / (1.0f + __expf(-x)); }

// ---------------- LSTM cell elementwise update ----------------
__global__ void lstm_cell(const float* __restrict__ gates,
                          float* __restrict__ c,
                          _Float16* __restrict__ hnext) {
    int idx = blockIdx.x * blockDim.x + threadIdx.x;
    if (idx >= M2 * HH) return;
    int m = idx >> 7, h = idx & (HH - 1);
    const float* g = gates + (size_t)m * G4;
    float ig = sigmoidf(g[h]);
    float fg = sigmoidf(g[HH + h]);
    float gg = tanhf(g[2 * HH + h]);
    float og = sigmoidf(g[3 * HH + h]);
    float cn = fg * c[idx] + ig * gg;
    c[idx] = cn;
    hnext[idx] = (_Float16)(og * tanhf(cn));
}

// ---------------- decoder: relu(h@W1+b1)@W2+b2, one block per row ----------------
__global__ void decoder(const _Float16* __restrict__ hf,
                        const float* __restrict__ w1, const float* __restrict__ b1,
                        const float* __restrict__ w2, const float* __restrict__ b2,
                        float* __restrict__ out) {
    __shared__ float hrow[HH];
    __shared__ float part[64];
    int m = blockIdx.x, j = threadIdx.x;     // 64 threads
    hrow[j]      = (float)hf[(size_t)m * HH + j];
    hrow[j + 64] = (float)hf[(size_t)m * HH + j + 64];
    __syncthreads();
    float acc = b1[j];
    for (int k = 0; k < HH; ++k) acc = fmaf(hrow[k], w1[k * 64 + j], acc);
    acc = fmaxf(acc, 0.0f);
    part[j] = acc * w2[j];
    __syncthreads();
    for (int s = 32; s > 0; s >>= 1) {
        if (j < s) part[j] += part[j + s];
        __syncthreads();
    }
    if (j == 0) out[m] = part[0] + b2[0];
}

// ---------------- host orchestration ----------------
extern "C" void kernel_launch(void* const* d_in, const int* in_sizes, int n_in,
                              void* d_out, int out_size, void* d_ws, size_t ws_size,
                              hipStream_t stream) {
    const float*      x    = (const float*)d_in[0];
    const long long*  ei   = (const long long*)d_in[1];
    const float*      g1w  = (const float*)d_in[2];
    const float*      g1b  = (const float*)d_in[3];
    const float*      g2w  = (const float*)d_in[4];
    const float*      g2b  = (const float*)d_in[5];
    const float*      wih  = (const float*)d_in[6];
    const float*      whh  = (const float*)d_in[7];
    const float*      bih  = (const float*)d_in[8];
    const float*      bhh  = (const float*)d_in[9];
    const float*      dw1  = (const float*)d_in[10];
    const float*      db1  = (const float*)d_in[11];
    const float*      dw2  = (const float*)d_in[12];
    const float*      db2  = (const float*)d_in[13];
    const int E = in_sizes[1] / 2;

    char* cur = (char*)d_ws;
    auto alloc = [&](size_t bytes) -> void* {
        void* p = (void*)cur;
        cur += (bytes + 255) & ~(size_t)255;
        return p;
    };
    float*     dinv  = (float*)alloc((size_t)NN * 4);
    float*     bufA  = (float*)alloc((size_t)BW * NN * HH * 4);    // f32 aggregation target
    float*     bufB  = (float*)alloc((size_t)BW * NN * HH * 4);    // f32 transform output
    _Float16*  af16  = (_Float16*)alloc((size_t)BW * NN * HH * 2); // f16 relu output (GEMM input)
    _Float16*  w2h   = (_Float16*)alloc((size_t)HH * HH * 2);
    _Float16*  xseq  = (_Float16*)alloc((size_t)M2 * WW * HH * 2);
    _Float16*  wcat  = (_Float16*)alloc((size_t)2 * HH * G4 * 2);
    float*     bcat  = (float*)alloc((size_t)G4 * 4);
    _Float16*  hbuf0 = (_Float16*)alloc((size_t)M2 * HH * 2);
    _Float16*  hbuf1 = (_Float16*)alloc((size_t)M2 * HH * 2);
    float*     cbuf  = (float*)alloc((size_t)M2 * HH * 4);
    float*     gates = (float*)alloc((size_t)M2 * G4 * 4);
    _Float16*  hb[2] = { hbuf0, hbuf1 };

    const int feat_total = BW * NN * HH;   // 20,480,000

    // degree (self-loop included via init 1.0) -> dinv
    fill_f32<<<(NN + 255) / 256, 256, 0, stream>>>(dinv, NN, 1.0f);
    degree_k<<<(E + 255) / 256, 256, 0, stream>>>(dinv, ei, E);
    rsqrt_k<<<(NN + 255) / 256, 256, 0, stream>>>(dinv, NN);

    // GCN layer 1: transform -> scatter -> bias+relu (emit f16)
    gcn1_transform<<<BW * NN, HH, 0, stream>>>(x, g1w, bufB);
    fill_f32<<<(feat_total + 255) / 256, 256, 0, stream>>>(bufA, feat_total, 0.0f);
    gcn_scatter<<<E + NN, HH, 0, stream>>>(bufB, bufA, ei, dinv, E);
    bias_relu_f16<<<(feat_total + 255) / 256, 256, 0, stream>>>(bufA, g1b, af16, feat_total);

    // GCN layer 2: WMMA GEMM (160000x128 @ 128x128, f16 operands) -> scatter -> bias+relu
    conv_f16<<<(HH * HH + 255) / 256, 256, 0, stream>>>(g2w, w2h, HH * HH);
    gcn2_gemm_wmma<<<BW * NN / 16, 256, 0, stream>>>(af16, w2h, bufB);
    fill_f32<<<(feat_total + 255) / 256, 256, 0, stream>>>(bufA, feat_total, 0.0f);
    gcn_scatter<<<E + NN, HH, 0, stream>>>(bufB, bufA, ei, dinv, E);
    bias_relu_f16<<<(feat_total + 255) / 256, 256, 0, stream>>>(bufA, g2b, af16, feat_total);

    // repack for LSTM, pack fused weights, init state
    build_xseq<<<(M2 * WW * HH + 255) / 256, 256, 0, stream>>>(af16, xseq);
    build_wcat<<<(2 * HH * G4 + 255) / 256, 256, 0, stream>>>(wih, whh, bih, bhh, wcat, bcat);
    fill_f16<<<(M2 * HH + 255) / 256, 256, 0, stream>>>(hbuf0, M2 * HH);
    fill_f32<<<(M2 * HH + 255) / 256, 256, 0, stream>>>(cbuf, M2 * HH, 0.0f);

    // 8 LSTM steps: fused [Xt|H] @ [Wih^T;Whh^T] WMMA GEMM + elementwise cell
    for (int t = 0; t < WW; ++t) {
        lstm_gates_wmma<<<dim3(M2 / 16, 4), 256, 0, stream>>>(
            xseq, hb[t & 1], wcat, bcat, gates, t);
        lstm_cell<<<(M2 * HH + 255) / 256, 256, 0, stream>>>(gates, cbuf, hb[(t + 1) & 1]);
    }

    // decoder (final hidden is in hb[0] after 8 steps)
    decoder<<<M2, 64, 0, stream>>>(hb[0], dw1, db1, dw2, db2, (float*)d_out);
}
